// VRWKV_ChannelMix_50955492000503
// MI455X (gfx1250) — matrix operand order repaired and checked by reading, not compiled, not run on hardware
//
#include <hip/hip_runtime.h>
#include <hip/hip_bf16.h>

typedef __attribute__((ext_vector_type(16))) _Float16 v16h;
typedef __attribute__((ext_vector_type(8)))  _Float16 v8h;
typedef __attribute__((ext_vector_type(8)))  float    v8f;

constexpr int B_   = 8;
constexpr int H_   = 128;
constexpr int W_   = 128;
constexpr int C_   = 192;
constexpr int HID_ = 768;
constexpr int T_   = H_ * W_;
constexpr int NT_  = B_ * T_;          // 131072 tokens
constexpr int MT_  = 32;               // tokens per gemm block
constexpr int KPAD = HID_ + 16;        // LDS row stride (halves)

// workspace layout (bytes)
constexpr size_t OFF_XK = 0;
constexpr size_t OFF_XR = OFF_XK + (size_t)NT_ * C_ * 2;
constexpr size_t OFF_WK = OFF_XR + (size_t)NT_ * C_ * 2;
constexpr size_t OFF_WV = OFF_WK + (size_t)HID_ * C_ * 2;
constexpr size_t OFF_WR = OFF_WV + (size_t)C_ * HID_ * 2;
constexpr size_t OFF_CB = OFF_WR + (size_t)C_ * C_ * 2;

// ---------------------------------------------------------------------------
// Fragment loaders. Layouts per CDNA5 ISA §7.12.2 (wave32, f16).
// A (16x32, MxK): lane<16 -> row M=lane, halves[0:8]=K[k0..k0+7],
//                 halves[8:16]=K[k0+16..k0+23]; lane>=16 shifts K by +8.
// B (32x16, KxN): lane<16 -> col N=lane, halves[0:8]=K[k0..k0+7],
//                 halves[8:16]=K[k0+16..k0+23]; lane>=16 shifts K by +8.
// Both map to contiguous 16B chunks of row-major [row][K] storage.
// ---------------------------------------------------------------------------
__device__ __forceinline__ v16h load_frag_A(const _Float16* __restrict__ base,
                                            int rowstride, int row0, int k0,
                                            int hp, int idx) {
  const _Float16* p = base + (size_t)(row0 + idx) * rowstride + k0 + hp * 8;
  v8h lo = *reinterpret_cast<const v8h*>(p);
  v8h hi = *reinterpret_cast<const v8h*>(p + 16);
  v16h a;
  #pragma unroll
  for (int i = 0; i < 8; ++i) { a[i] = lo[i]; a[8 + i] = hi[i]; }
  return a;
}

__device__ __forceinline__ v16h load_frag_A_lds(const _Float16* p_row,
                                                int k0, int hp) {
  const _Float16* p = p_row + k0 + hp * 8;
  v8h lo = *reinterpret_cast<const v8h*>(p);
  v8h hi = *reinterpret_cast<const v8h*>(p + 16);
  v16h a;
  #pragma unroll
  for (int i = 0; i < 8; ++i) { a[i] = lo[i]; a[8 + i] = hi[i]; }
  return a;
}

__device__ __forceinline__ v16h load_frag_B(const _Float16* __restrict__ base,
                                            int rowstride, int n0, int k0,
                                            int hp, int idx) {
  const _Float16* p = base + (size_t)(n0 + idx) * rowstride + k0 + hp * 8;
  v8h lo = *reinterpret_cast<const v8h*>(p);
  v8h hi = *reinterpret_cast<const v8h*>(p + 16);
  v16h b;
  #pragma unroll
  for (int i = 0; i < 8; ++i) { b[i] = lo[i]; b[8 + i] = hi[i]; }
  return b;
}

#define WMMA_F16(a, b, c) \
  __builtin_amdgcn_wmma_f32_16x16x32_f16(false, (a), false, (b), (short)0, (c), false, false)

// ---------------------------------------------------------------------------
// Kernel 1: weight conversion (f32 -> f16) + fold OmniShift into one 5x5
// per-channel stencil:  comb[tap][c] = a3*w5 + a2*w3(center 3x3) +
//                                      (a0 + a1*w1) at the center tap.
// ---------------------------------------------------------------------------
__global__ __launch_bounds__(256) void prep_kernel(
    const float* __restrict__ Wk, const float* __restrict__ Wv,
    const float* __restrict__ Wr, const float* __restrict__ w1,
    const float* __restrict__ w3, const float* __restrict__ w5,
    const float* __restrict__ alpha,
    _Float16* __restrict__ Wk_h, _Float16* __restrict__ Wv_h,
    _Float16* __restrict__ Wr_h, float* __restrict__ comb) {
  int tid = blockIdx.x * blockDim.x + threadIdx.x;
  if (tid < HID_ * C_) Wk_h[tid] = (_Float16)Wk[tid];
  if (tid < C_ * HID_) Wv_h[tid] = (_Float16)Wv[tid];
  if (tid < C_ * C_)   Wr_h[tid] = (_Float16)Wr[tid];
  if (tid < C_) {
    int c = tid;
    float a0 = alpha[0], a1 = alpha[1], a2 = alpha[2], a3 = alpha[3];
    #pragma unroll
    for (int ky = 0; ky < 5; ++ky) {
      #pragma unroll
      for (int kx = 0; kx < 5; ++kx) {
        float cf = a3 * w5[c * 25 + ky * 5 + kx];
        if (ky >= 1 && ky <= 3 && kx >= 1 && kx <= 3)
          cf += a2 * w3[c * 9 + (ky - 1) * 3 + (kx - 1)];
        if (ky == 2 && kx == 2) cf += a0 + a1 * w1[c];
        comb[(ky * 5 + kx) * C_ + c] = cf;
      }
    }
  }
}

// ---------------------------------------------------------------------------
// Kernel 2: OmniShift stencil + token-mix, emitting f16 xk/xr in [t][C]
// row-major (== WMMA A-fragment friendly). One block per (b, h) image row,
// 192 threads = channels (fully coalesced 768B row segments, 5-row working
// set ~0.5 MB stays in cache; whole x (100 MB) fits in the 192 MB L2).
// ---------------------------------------------------------------------------
__global__ __launch_bounds__(192) void omni_mix_kernel(
    const float* __restrict__ x, const float* __restrict__ comb,
    const float* __restrict__ mix_k, const float* __restrict__ mix_r,
    _Float16* __restrict__ xk, _Float16* __restrict__ xr) {
  int c = threadIdx.x;
  int b = blockIdx.x / H_;
  int h = blockIdx.x % H_;
  float mk = mix_k[c], mr = mix_r[c];
  float cf[25];
  #pragma unroll
  for (int j = 0; j < 25; ++j) cf[j] = comb[j * C_ + c];
  size_t base = (size_t)b * T_ * C_;
  for (int w = 0; w < W_; ++w) {
    float xc  = x[base + (size_t)(h * W_ + w) * C_ + c];
    float acc = 0.f;
    #pragma unroll
    for (int ky = 0; ky < 5; ++ky) {
      int hh = h + ky - 2;
      if (hh < 0 || hh >= H_) continue;
      #pragma unroll
      for (int kx = 0; kx < 5; ++kx) {
        int ww = w + kx - 2;
        if (ww < 0 || ww >= W_) continue;
        acc += cf[ky * 5 + kx] * x[base + (size_t)(hh * W_ + ww) * C_ + c];
      }
    }
    size_t t = (size_t)b * T_ + h * W_ + w;
    xk[t * C_ + c] = (_Float16)(xc * mk + acc * (1.f - mk));
    xr[t * C_ + c] = (_Float16)(xc * mr + acc * (1.f - mr));
  }
}

// ---------------------------------------------------------------------------
// Kernel 3: fused GEMM pipeline per 32-token block (8 wave32 waves).
//   k = relu(xk Wk^T)^2  -> LDS (f16, 32x784)   [K=192]
//   layernorm(k) in-place (wave-parallel, __shfl_xor reductions)
//   out = sigmoid(xr Wr^T) * (k Wv^T)           [K=768 / K=192]
// Wave owns a fixed 16-row M-tile; A fragments hoisted to registers and
// reused across all N-tiles; multiple independent accumulators per wave
// for back-to-back WMMA issue. `k` never touches HBM.
// ---------------------------------------------------------------------------
__global__ __launch_bounds__(256) void gemm_kernel(
    const _Float16* __restrict__ xk, const _Float16* __restrict__ xr,
    const _Float16* __restrict__ Wk_h, const _Float16* __restrict__ Wv_h,
    const _Float16* __restrict__ Wr_h,
    const float* __restrict__ ln_g, const float* __restrict__ ln_b,
    float* __restrict__ out) {
  __shared__ _Float16 ksh[MT_ * KPAD];  // 32*784*2 = 50176 B

  const int wave = threadIdx.x >> 5;
  const int lane = threadIdx.x & 31;
  const int hp   = lane >> 4;     // which half of the wave
  const int idx  = lane & 15;
  const int t0   = blockIdx.x * MT_;
  const int mt   = wave >> 2;     // wave's M-tile (0..1)

  // ---- GEMM1: k_raw = relu(xk . Wk^T)^2 ----
  // A fragments (6 K-steps) hoisted once; 12 N-tiles per wave in groups of 4.
  {
    v16h afr[6];
    #pragma unroll
    for (int s = 0; s < 6; ++s)
      afr[s] = load_frag_A(xk, C_, t0 + mt * 16, s * 32, hp, idx);

    const int nt0 = (wave & 3) * 12;
    #pragma unroll
    for (int g = 0; g < 3; ++g) {
      v8f acc[4] = {{}, {}, {}, {}};
      #pragma unroll
      for (int s = 0; s < 6; ++s) {
        #pragma unroll
        for (int u = 0; u < 4; ++u) {
          v16h b = load_frag_B(Wk_h, C_, (nt0 + g * 4 + u) * 16, s * 32, hp, idx);
          acc[u] = WMMA_F16(afr[s], b, acc[u]);
        }
      }
      #pragma unroll
      for (int u = 0; u < 4; ++u) {
        int nt = nt0 + g * 4 + u;
        #pragma unroll
        for (int j = 0; j < 8; ++j) {
          float v = acc[u][j];
          v = v > 0.f ? v * v : 0.f;
          ksh[(mt * 16 + hp * 8 + j) * KPAD + nt * 16 + idx] = (_Float16)v;
        }
      }
    }
  }
  __syncthreads();

  // ---- LayerNorm over HID per token row (4 rows per wave) ----
  for (int r = wave * 4; r < wave * 4 + 4; ++r) {
    float s = 0.f, ss = 0.f;
    for (int j = lane; j < HID_; j += 32) {
      float v = (float)ksh[r * KPAD + j];
      s += v; ss += v * v;
    }
    #pragma unroll
    for (int m = 16; m >= 1; m >>= 1) {
      s  += __shfl_xor(s,  m, 32);
      ss += __shfl_xor(ss, m, 32);
    }
    float mu   = s / HID_;
    float var  = ss / HID_ - mu * mu;
    float rstd = rsqrtf(var + 1e-5f);
    for (int j = lane; j < HID_; j += 32) {
      float v = (float)ksh[r * KPAD + j];
      ksh[r * KPAD + j] = (_Float16)((v - mu) * rstd * ln_g[j] + ln_b[j]);
    }
  }
  __syncthreads();

  // ---- GEMM2 (kv = k . Wv^T) + GEMM3 (r = sigmoid(xr . Wr^T)), fused ----
  // 3 N-tiles per wave, 3 live accumulators; LDS A fragment loaded once per
  // K-step and reused by all 3 WMMAs.
  {
    const int nt0 = (wave & 3) * 3;
    const _Float16* krow = &ksh[(mt * 16 + idx) * KPAD];

    v8f ckv[3] = {{}, {}, {}};
    #pragma unroll 6
    for (int kk = 0; kk < HID_; kk += 32) {
      v16h a = load_frag_A_lds(krow, kk, hp);
      #pragma unroll
      for (int u = 0; u < 3; ++u) {
        v16h b = load_frag_B(Wv_h, HID_, (nt0 + u) * 16, kk, hp, idx);
        ckv[u] = WMMA_F16(a, b, ckv[u]);
      }
    }

    v8f cr[3] = {{}, {}, {}};
    {
      v16h xrfr[6];
      #pragma unroll
      for (int s = 0; s < 6; ++s)
        xrfr[s] = load_frag_A(xr, C_, t0 + mt * 16, s * 32, hp, idx);
      #pragma unroll
      for (int s = 0; s < 6; ++s) {
        #pragma unroll
        for (int u = 0; u < 3; ++u) {
          v16h b = load_frag_B(Wr_h, C_, (nt0 + u) * 16, s * 32, hp, idx);
          cr[u] = WMMA_F16(xrfr[s], b, cr[u]);
        }
      }
    }

    #pragma unroll
    for (int u = 0; u < 3; ++u) {
      int nt = nt0 + u;
      #pragma unroll
      for (int j = 0; j < 8; ++j) {
        float rv = 1.f / (1.f + __expf(-cr[u][j]));
        size_t row = (size_t)t0 + mt * 16 + hp * 8 + j;
        out[row * C_ + nt * 16 + idx] = rv * ckv[u][j];
      }
    }
  }
}

// ---------------------------------------------------------------------------
extern "C" void kernel_launch(void* const* d_in, const int* in_sizes, int n_in,
                              void* d_out, int out_size, void* d_ws, size_t ws_size,
                              hipStream_t stream) {
  const float* x     = (const float*)d_in[0];
  const float* w1    = (const float*)d_in[1];
  const float* w3    = (const float*)d_in[2];
  const float* w5    = (const float*)d_in[3];
  const float* alpha = (const float*)d_in[4];
  const float* mix_k = (const float*)d_in[5];
  const float* mix_r = (const float*)d_in[6];
  const float* Wk    = (const float*)d_in[7];
  const float* ln_g  = (const float*)d_in[8];
  const float* ln_b  = (const float*)d_in[9];
  const float* Wr    = (const float*)d_in[10];
  const float* Wv    = (const float*)d_in[11];
  float* out = (float*)d_out;

  char* ws = (char*)d_ws;
  _Float16* xk_h = (_Float16*)(ws + OFF_XK);
  _Float16* xr_h = (_Float16*)(ws + OFF_XR);
  _Float16* Wk_h = (_Float16*)(ws + OFF_WK);
  _Float16* Wv_h = (_Float16*)(ws + OFF_WV);
  _Float16* Wr_h = (_Float16*)(ws + OFF_WR);
  float*    comb = (float*)   (ws + OFF_CB);

  int prep_blocks = (HID_ * C_ + 255) / 256;
  prep_kernel<<<prep_blocks, 256, 0, stream>>>(Wk, Wv, Wr, w1, w3, w5, alpha,
                                               Wk_h, Wv_h, Wr_h, comb);
  omni_mix_kernel<<<B_ * H_, 192, 0, stream>>>(x, comb, mix_k, mix_r, xk_h, xr_h);
  gemm_kernel<<<NT_ / MT_, 256, 0, stream>>>(xk_h, xr_h, Wk_h, Wv_h, Wr_h,
                                             ln_g, ln_b, out);
}
